// PureAttention1D_31834297598709
// MI455X (gfx1250) — compile-verified, matrix operand order checked by reading, not run on hardware
//
#include <hip/hip_runtime.h>
#include <math.h>

typedef _Float16 h16;
typedef __attribute__((ext_vector_type(16))) _Float16 v16h;
typedef __attribute__((ext_vector_type(8)))  float    v8f;
typedef __attribute__((ext_vector_type(4)))  unsigned v4u;
typedef __attribute__((ext_vector_type(8)))  unsigned v8u;

#define BATCH 4
#define NSEQ  4096
#define DDIM  512
#define BM    64          // queries per workgroup
#define BN    64          // keys per iteration
#define QS_STRIDE 520     // DDIM + 8 elements -> 1040B rows, 16B aligned, bank-skewed
#define KS_STRIDE 520     // matches TDM pad: 256 dwords data + 4 dwords pad per row
#define VT_STRIDE 72      // BN + 8
#define SB_STRIDE 68      // BN + 4 (f32)
#define PB_STRIDE 72
#define ATTN_SCALE 0.04419417382415922f   // 1/sqrt(512)
#define LOG2E      1.4426950408889634f

struct __align__(16) FaShared {
  h16   Qs[BM * QS_STRIDE];    // Q tile, row-major [q][d]
  h16   Ks[BN * KS_STRIDE];    // K tile, row-major [key][d]  (B operand of S = Q*K^T)
  h16   Vt[DDIM * VT_STRIDE];  // V tile, transposed [d][key] (B operand of O += P*V)
  float Sb[BM * SB_STRIDE];    // raw scores tile (f32)
  h16   Pb[BM * PB_STRIDE];    // softmaxed probs (f16)
  float rowM[BM];
  float rowL[BM];
  float rowAlpha[BM];
  float rowInvL[BM];
};

// Low 32 bits of a generic shared pointer == LDS byte offset (ISA 10.2 aperture rule).
__device__ __forceinline__ unsigned lds_off(const void* p) {
  return (unsigned)(uintptr_t)p;
}

// 16x32 f16 fragment (A layout; also serves B when memory layout is [n][k] row-major).
// Lane L: row = L&15, half = L>>4. VGPR0..3: k = half*8+0..7, VGPR4..7: k = 16+half*8+0..7.
__device__ __forceinline__ v16h load_frag(const h16* base, int row0, int stride,
                                          int k0, int lane) {
  const int r  = lane & 15;
  const int hf = lane >> 4;
  const h16* p = base + (row0 + r) * stride + k0 + hf * 8;
  union { uint4 u[2]; v16h v; } t;
  t.u[0] = *(const uint4*)(p);
  t.u[1] = *(const uint4*)(p + 16);
  return t.v;
}

__global__ void __launch_bounds__(256, 1)
fa_convert_f32_to_f16(const float* __restrict__ x, h16* __restrict__ xh, int n) {
  int i = (blockIdx.x * 256 + threadIdx.x) * 4;
  if (i < n) {
    float4 v = *(const float4*)(x + i);
    h16 o[4] = {(h16)v.x, (h16)v.y, (h16)v.z, (h16)v.w};
    *(uint2*)(xh + i) = *(const uint2*)o;
  }
}

__global__ void __launch_bounds__(256, 1)
fa_flash_attn(const h16* __restrict__ Xh, float* __restrict__ Out) {
  __shared__ FaShared sm;
  const int tid  = threadIdx.x;
  const int lane = tid & 31;
  const int wave = tid >> 5;
  const int nqt  = NSEQ / BM;
  const int b    = blockIdx.x / nqt;
  const int q0   = (blockIdx.x % nqt) * BM;
  const h16* Xb  = Xh + (size_t)b * NSEQ * DDIM;

  // ---- Q tile: async global->LDS b128 copies (ASYNCcnt path, no VGPR roundtrip) ----
  {
    const unsigned qs_base = lds_off(&sm.Qs[0]);
    #pragma unroll
    for (int i = 0; i < 16; ++i) {
      int idx = tid + i * 256;               // 4096 16B chunks
      int r   = idx >> 6;
      int c8  = idx & 63;
      const h16* g = Xb + (size_t)(q0 + r) * DDIM + c8 * 8;
      unsigned  l  = qs_base + (unsigned)((r * QS_STRIDE + c8 * 8) * 2);
      asm volatile("global_load_async_to_lds_b128 %0, %1, off"
                   :: "v"(l), "v"(g) : "memory");
    }
  }
  if (tid < BM) {
    sm.rowM[tid] = -__builtin_huge_valf();
    sm.rowL[tid] = 0.0f;
  }

  v8f accO[16];               // wave's 64q x 64d output slab, f32
  #pragma unroll
  for (int i = 0; i < 16; ++i) accO[i] = (v8f){};

  asm volatile("s_wait_asynccnt 0x0" ::: "memory");
  __syncthreads();

  const int tr = wave >> 1;           // GEMM1: query tile row (0..3)
  const int c0 = (wave & 1) * 32;     // GEMM1: key column base (0 or 32)
  const int hf = lane >> 4;
  const int nn = lane & 15;
  const unsigned ks_lds = lds_off(&sm.Ks[0]);

  for (int k0 = 0; k0 < NSEQ; k0 += BN) {
    // ---- K tile via Tensor Data Mover: one 2D 64x512 f16 tile, padded rows ----
    if (wave == 0) {
      unsigned long long ga =
          (unsigned long long)(uintptr_t)(Xb + (size_t)k0 * DDIM);
      v4u g0;
      g0[0] = 1u;                                    // count=1, user descriptor
      g0[1] = ks_lds;                                // lds_addr  [63:32]
      g0[2] = (unsigned)(ga & 0xffffffffu);          // global_addr low  [95:64]
      g0[3] = (unsigned)((ga >> 32) & 0x01ffffffu)   // global_addr high [120:96]
            | (2u << 30);                            // type = 2 (image)
      v8u g1;
      g1[0] = (1u << 16)       // data_size = 2 bytes
            | (1u << 20)       // pad_enable
            | (7u << 22)       // pad_interval: every 256 dwords (= one 512-elem row)
            | (3u << 25);      // pad_amount: 4 dwords -> 520-element row stride
      g1[1] = (512u & 0xffffu) << 16;                // tensor_dim0[15:0]
      g1[2] = (512u >> 16) | ((4096u & 0xffffu) << 16); // dim0 hi | tensor_dim1 lo
      g1[3] = (4096u >> 16) | (512u << 16);          // dim1 hi | tile_dim0 = 512
      g1[4] = 64u;                                   // tile_dim1 = 64 keys, tile_dim2 = 0
      g1[5] = 512u;                                  // tensor_dim0_stride low
      g1[6] = 0u;                                    // dim0_stride hi | dim1_stride lo
      g1[7] = 0u;
      v4u gz0 = (v4u){0u, 0u, 0u, 0u};
      v4u gz1 = (v4u){0u, 0u, 0u, 0u};
      asm volatile("tensor_load_to_lds %0, %1, %2, %3"
                   :: "s"(g0), "s"(g1), "s"(gz0), "s"(gz1) : "memory");
    }

    // ---- V tile: register-blocked transpose (4 keys x 8 d) -> b64 LDS stores ----
    #pragma unroll
    for (int i = 0; i < 4; ++i) {
      int t  = tid + i * 256;        // 1024 tiles: 16 key-groups x 64 d-chunks
      int kg = t >> 6;
      int c8 = t & 63;
      const h16* src = Xb + (size_t)(k0 + 4 * kg) * DDIM + c8 * 8;
      union { uint4 u; h16 h[8]; } a0, a1, a2, a3;
      a0.u = *(const uint4*)(src);
      a1.u = *(const uint4*)(src + DDIM);
      a2.u = *(const uint4*)(src + 2 * DDIM);
      a3.u = *(const uint4*)(src + 3 * DDIM);
      #pragma unroll
      for (int e = 0; e < 8; ++e) {
        union { uint2 u; h16 h[4]; } pk;
        pk.h[0] = a0.h[e]; pk.h[1] = a1.h[e]; pk.h[2] = a2.h[e]; pk.h[3] = a3.h[e];
        *(uint2*)(&sm.Vt[(c8 * 8 + e) * VT_STRIDE + 4 * kg]) = pk.u;
      }
    }
    if (wave == 0) __builtin_amdgcn_s_wait_tensorcnt(0);
    __syncthreads();

    // ---- GEMM1: S[16tr..+16, c0..+32] = Q * K^T ----
    v8f acc0 = (v8f){}, acc1 = (v8f){};
    #pragma unroll
    for (int kk = 0; kk < DDIM; kk += 32) {
      v16h a  = load_frag(sm.Qs, 16 * tr, QS_STRIDE, kk, lane);
      v16h b0 = load_frag(sm.Ks, c0,      KS_STRIDE, kk, lane);
      v16h b1 = load_frag(sm.Ks, c0 + 16, KS_STRIDE, kk, lane);
      acc0 = __builtin_amdgcn_wmma_f32_16x16x32_f16(false, a, false, b0,
                                                    (short)0, acc0, false, false);
      acc1 = __builtin_amdgcn_wmma_f32_16x16x32_f16(false, a, false, b1,
                                                    (short)0, acc1, false, false);
    }
    #pragma unroll
    for (int e = 0; e < 8; ++e) {
      int m = 16 * tr + hf * 8 + e;
      sm.Sb[m * SB_STRIDE + c0 + nn]      = acc0[e];
      sm.Sb[m * SB_STRIDE + c0 + 16 + nn] = acc1[e];
    }
    __syncthreads();

    // ---- online softmax: one thread per query row ----
    if (tid < BM) {
      const int row = tid;
      float mOld = sm.rowM[row];
      float mNew = mOld;
      #pragma unroll 8
      for (int j = 0; j < BN; ++j)
        mNew = fmaxf(mNew, sm.Sb[row * SB_STRIDE + j] * ATTN_SCALE);
      float alpha = exp2f((mOld - mNew) * LOG2E);
      float l = sm.rowL[row] * alpha;
      #pragma unroll 8
      for (int j = 0; j < BN; ++j) {
        float p = exp2f((sm.Sb[row * SB_STRIDE + j] * ATTN_SCALE - mNew) * LOG2E);
        l += p;
        sm.Pb[row * PB_STRIDE + j] = (h16)p;
      }
      sm.rowM[row]     = mNew;
      sm.rowL[row]     = l;
      sm.rowAlpha[row] = alpha;
    }
    __syncthreads();

    // ---- rescale O accumulators by alpha ----
    #pragma unroll
    for (int tm = 0; tm < 4; ++tm) {
      float al[8];
      #pragma unroll
      for (int e = 0; e < 8; ++e) al[e] = sm.rowAlpha[16 * tm + hf * 8 + e];
      #pragma unroll
      for (int tc = 0; tc < 4; ++tc) {
        #pragma unroll
        for (int e = 0; e < 8; ++e) accO[tm * 4 + tc][e] *= al[e];
      }
    }

    // ---- GEMM2: O[64q x 64d slab] += P * V ----
    #pragma unroll
    for (int kk = 0; kk < BN; kk += 32) {
      v16h bfr[4];
      #pragma unroll
      for (int tc = 0; tc < 4; ++tc)
        bfr[tc] = load_frag(sm.Vt, wave * 64 + 16 * tc, VT_STRIDE, kk, lane);
      #pragma unroll
      for (int tm = 0; tm < 4; ++tm) {
        v16h a = load_frag(sm.Pb, 16 * tm, PB_STRIDE, kk, lane);
        #pragma unroll
        for (int tc = 0; tc < 4; ++tc)
          accO[tm * 4 + tc] = __builtin_amdgcn_wmma_f32_16x16x32_f16(
              false, a, false, bfr[tc], (short)0, accO[tm * 4 + tc], false, false);
      }
    }
    __syncthreads();   // protect Ks/Vt/Sb before next stage
  }

  if (tid < BM) sm.rowInvL[tid] = 1.0f / sm.rowL[tid];
  __syncthreads();

  // ---- epilogue: divide by l, store f32 ----
  #pragma unroll
  for (int tm = 0; tm < 4; ++tm) {
    float inv[8];
    #pragma unroll
    for (int e = 0; e < 8; ++e) inv[e] = sm.rowInvL[16 * tm + hf * 8 + e];
    #pragma unroll
    for (int tc = 0; tc < 4; ++tc) {
      #pragma unroll
      for (int e = 0; e < 8; ++e) {
        int m = 16 * tm + hf * 8 + e;
        int d = wave * 64 + 16 * tc + nn;
        Out[((size_t)b * NSEQ + q0 + m) * DDIM + d] = accO[tm * 4 + tc][e] * inv[e];
      }
    }
  }
}

extern "C" void kernel_launch(void* const* d_in, const int* in_sizes, int n_in,
                              void* d_out, int out_size, void* d_ws, size_t ws_size,
                              hipStream_t stream) {
  const float* x = (const float*)d_in[0];
  float* out = (float*)d_out;
  h16* xh = (h16*)d_ws;   // 4*4096*512 f16 = 16 MB scratch
  const int total = BATCH * NSEQ * DDIM;

  fa_convert_f32_to_f16<<<total / 4 / 256, 256, 0, stream>>>(x, xh, total);
  fa_flash_attn<<<BATCH * (NSEQ / BM), 256, 0, stream>>>(xh, out);
}
